// MultiHeadAttention_86045374808101
// MI455X (gfx1250) — compile-verified, matrix operand order checked by reading, not run on hardware
//
#include <hip/hip_runtime.h>

#define D_MODEL 1024
#define N_HEAD  16
#define D_HEAD  64
#define BATCH   4
#define SEQ     2048
#define M_TOTAL (BATCH * SEQ)   // 8192
#define N_QKV   (3 * D_MODEL)   // 3072

typedef __attribute__((ext_vector_type(8)))  _Float16 v8h;
typedef __attribute__((ext_vector_type(16))) _Float16 v16h;
typedef __attribute__((ext_vector_type(8)))  float    v8f;
typedef __attribute__((ext_vector_type(4)))  int      v4i;

#define AS_GLOBAL __attribute__((address_space(1)))
#define AS_LDS    __attribute__((address_space(3)))

// ---------------------------------------------------------------------------
// CDNA5 async global->LDS copy (GLOBAL_LOAD_ASYNC_TO_LDS_B128, ASYNCcnt),
// with sync fallback when the builtin is unavailable.
// ---------------------------------------------------------------------------
#if defined(__has_builtin)
#if __has_builtin(__builtin_amdgcn_global_load_async_to_lds_b128) && \
    __has_builtin(__builtin_amdgcn_s_wait_asynccnt)
#define HAVE_ASYNC_LDS 1
#endif
#endif

__device__ __forceinline__ void copy16_g2l(_Float16* ldst,
                                           const _Float16* gsrc) {
#ifdef HAVE_ASYNC_LDS
  __builtin_amdgcn_global_load_async_to_lds_b128(
      (AS_GLOBAL v4i*)(void*)gsrc,
      (AS_LDS    v4i*)(void*)ldst, 0, 0);
#else
  *(v8h*)ldst = *(const v8h*)gsrc;
#endif
}

__device__ __forceinline__ void async_wait_all() {
#ifdef HAVE_ASYNC_LDS
  __builtin_amdgcn_s_wait_asynccnt(0);
#endif
}

// ---------------------------------------------------------------------------
// WMMA helpers (CDNA5: V_WMMA_F32_16X16X32_F16, wave32)
// ---------------------------------------------------------------------------
__device__ __forceinline__ v8f wmma_f16(v16h a, v16h b, v8f c) {
  return __builtin_amdgcn_wmma_f32_16x16x32_f16(false, a, false, b, (short)0, c,
                                                false, false);
}

// 16x32 f16 A-style fragment (ISA 7.12.2): lane L holds row M = L&15;
// kbase = (L<16 ? 0 : 8); VGPR0..3 = K kbase+0..7, VGPR4..7 = K kbase+16..23
// -> two contiguous 16-byte loads per lane.
__device__ __forceinline__ v16h frag_ld(const _Float16* rowbase, int kofs) {
  v8h lo = *(const v8h*)(rowbase + kofs);
  v8h hi = *(const v8h*)(rowbase + kofs + 16);
  v16h r;
#pragma unroll
  for (int i = 0; i < 8; ++i) { r[i] = lo[i]; r[i + 8] = hi[i]; }
  return r;
}

// ---------------------------------------------------------------------------
// fp32 -> fp16 cast, 8 elements/thread
// ---------------------------------------------------------------------------
__global__ void cast_f32_to_f16_v8(const float* __restrict__ src,
                                   _Float16* __restrict__ dst, int n8) {
  int i = blockIdx.x * blockDim.x + threadIdx.x;
  if (i < n8) {
    const float4* s = (const float4*)src + (size_t)i * 2;
    float4 a = s[0], b = s[1];
    v8h o;
    o[0] = (_Float16)a.x; o[1] = (_Float16)a.y;
    o[2] = (_Float16)a.z; o[3] = (_Float16)a.w;
    o[4] = (_Float16)b.x; o[5] = (_Float16)b.y;
    o[6] = (_Float16)b.z; o[7] = (_Float16)b.w;
    *((v8h*)dst + i) = o;
  }
}

// ---------------------------------------------------------------------------
// fp32 [rows][cols] -> fp16 transposed [cols][rows] (32x32 LDS tiles)
// ---------------------------------------------------------------------------
__global__ void cast_transpose_f16(const float* __restrict__ src,
                                   _Float16* __restrict__ dst,
                                   int rows, int cols) {
  __shared__ _Float16 t[32][33];
  const int c0 = blockIdx.x * 32, r0 = blockIdx.y * 32;
#pragma unroll
  for (int i = 0; i < 32; i += 8) {
    int r = r0 + threadIdx.y + i, c = c0 + threadIdx.x;
    t[threadIdx.y + i][threadIdx.x] = (_Float16)src[(size_t)r * cols + c];
  }
  __syncthreads();
#pragma unroll
  for (int i = 0; i < 32; i += 8) {
    int r = c0 + threadIdx.y + i;   // output row = source col
    int c = r0 + threadIdx.x;       // output col = source row
    dst[(size_t)r * rows + c] = t[threadIdx.x][threadIdx.y + i];
  }
}

// ---------------------------------------------------------------------------
// GEMM core: 128x128x64 block tile, double-buffered LDS, async copies.
// 16 WMMA per wave per barrier.  Both A [M][K] and Bt [N][K] are
// K-contiguous -> identical b128 copy paths.
// ---------------------------------------------------------------------------
#define BM 128
#define BN 128
#define BK 64
#define LDT 72   // LDS row stride in f16 (144B, 16B-aligned, bank-staggered)

// ---------------------------------------------------------------------------
// QKV projection + scatter into Q [B,H,T,Dh], K [B,H,T,Dh], V [B,H,Dh,T]
// ---------------------------------------------------------------------------
__launch_bounds__(256)
__global__ void qkv_gemm(const _Float16* __restrict__ X,   // [8192][1024]
                         const _Float16* __restrict__ Wt,  // [3072][1024]
                         const float*    __restrict__ bias,// [3072]
                         _Float16* __restrict__ q,
                         _Float16* __restrict__ k,
                         _Float16* __restrict__ v) {
  __shared__ _Float16 sA[2][BM * LDT];
  __shared__ _Float16 sB[2][BN * LDT];

  const int tid  = threadIdx.x;
  const int lane = tid & 31;
  const int wid  = tid >> 5;
  const int wm   = wid & 3;
  const int wn   = wid >> 2;
  const int m0   = blockIdx.y * BM;
  const int n0   = blockIdx.x * BN;
  const int lrow = lane & 15;
  const int kofs = (lane < 16) ? 0 : 8;
  const int half8 = (lane >> 4) << 3;

  const int cr = tid >> 1;             // copy row 0..127
  const int cs = (tid & 1) * 32;       // copy segment (32 f16 per thread)
  auto load_tiles = [&](int buf, int k0) {
    const _Float16* ga = X + (size_t)(m0 + cr) * D_MODEL + k0 + cs;
    _Float16* la = &sA[buf][cr * LDT + cs];
#pragma unroll
    for (int p = 0; p < 4; ++p) copy16_g2l(la + p * 8, ga + p * 8);
    const _Float16* gb = Wt + (size_t)(n0 + cr) * D_MODEL + k0 + cs;
    _Float16* lb = &sB[buf][cr * LDT + cs];
#pragma unroll
    for (int p = 0; p < 4; ++p) copy16_g2l(lb + p * 8, gb + p * 8);
  };

  v8f acc[2][4];
#pragma unroll
  for (int i = 0; i < 2; ++i)
#pragma unroll
    for (int j = 0; j < 4; ++j) acc[i][j] = (v8f){0,0,0,0,0,0,0,0};

  load_tiles(0, 0);
  async_wait_all();
  __syncthreads();

  const int NK = D_MODEL / BK;   // 16
  for (int kk = 0; kk < NK; ++kk) {
    const int cur = kk & 1;
    if (kk + 1 < NK) load_tiles(cur ^ 1, (kk + 1) * BK);

#pragma unroll
    for (int kc = 0; kc < 2; ++kc) {
      const int ko = kc * 32 + kofs;
      v16h af0 = frag_ld(&sA[cur][(wm * 32 +      lrow) * LDT], ko);
      v16h af1 = frag_ld(&sA[cur][(wm * 32 + 16 + lrow) * LDT], ko);
#pragma unroll
      for (int j = 0; j < 4; ++j) {
        v16h bf = frag_ld(&sB[cur][(wn * 64 + j * 16 + lrow) * LDT], ko);
        acc[0][j] = wmma_f16(af0, bf, acc[0][j]);
        acc[1][j] = wmma_f16(af1, bf, acc[1][j]);
      }
    }
    async_wait_all();
    __syncthreads();
  }

#pragma unroll
  for (int i = 0; i < 2; ++i)
#pragma unroll
    for (int j = 0; j < 4; ++j)
#pragma unroll
      for (int r = 0; r < 8; ++r) {
        int m = m0 + wm * 32 + i * 16 + r + half8;
        int c = n0 + wn * 64 + j * 16 + lrow;
        float val = acc[i][j][r] + bias[c];
        int sel = c >> 10;
        int rem = c & 1023;
        int h   = rem >> 6;
        int dh  = rem & 63;
        int b   = m >> 11;
        int t   = m & 2047;
        if (sel == 0) {
          q[(((size_t)b * N_HEAD + h) * SEQ + t) * D_HEAD + dh] = (_Float16)val;
        } else if (sel == 1) {
          k[(((size_t)b * N_HEAD + h) * SEQ + t) * D_HEAD + dh] = (_Float16)val;
        } else {
          // V stored transposed: [B,H,Dh,T]
          v[(((size_t)b * N_HEAD + h) * D_HEAD + dh) * SEQ + t] = (_Float16)val;
        }
      }
}

// ---------------------------------------------------------------------------
// Flash attention: block = (b,h) x 128 query rows; wave = 16-row tile.
// K tiles [key][dh] and V tiles [dh][key] both copy contiguously (V is
// pre-transposed in global).  Double-buffered chunks, online softmax.
// ---------------------------------------------------------------------------
#define SK  64
#define LDK 72   // 144B row stride

__launch_bounds__(256)
__global__ void flash_attn(const _Float16* __restrict__ Q,   // [B,H,T,Dh]
                           const _Float16* __restrict__ K,   // [B,H,T,Dh]
                           const _Float16* __restrict__ V,   // [B,H,Dh,T]
                           _Float16* __restrict__ O) {        // [B,T,C] f16
  __shared__ _Float16 sK[2][SK * LDK];
  __shared__ _Float16 sVt[2][D_HEAD * LDK];
  __shared__ _Float16 sP[8 * 16 * LDK];

  const int tid  = threadIdx.x;
  const int lane = tid & 31;
  const int wid  = tid >> 5;
  const int bh   = blockIdx.y;
  const int q0   = blockIdx.x * 128 + wid * 16;
  const int lrow = lane & 15;
  const int kofs = (lane < 16) ? 0 : 8;
  const int half8 = (lane >> 4) << 3;

  const _Float16* qbh = Q + (size_t)bh * SEQ * D_HEAD;
  const _Float16* kbh = K + (size_t)bh * SEQ * D_HEAD;
  const _Float16* vbh = V + (size_t)bh * SEQ * D_HEAD;  // [Dh][T]

  const int cr = tid >> 2;            // 0..63
  const int cs = (tid & 3) * 16;      // 0/16/32/48
  auto load_chunk = [&](int buf, int c0) {
    const _Float16* gk = kbh + (size_t)(c0 + cr) * D_HEAD + cs;
    copy16_g2l(&sK[buf][cr * LDK + cs], gk);
    copy16_g2l(&sK[buf][cr * LDK + cs + 8], gk + 8);
    const _Float16* gv = vbh + (size_t)cr * SEQ + c0 + cs;  // dh row, keys
    copy16_g2l(&sVt[buf][cr * LDK + cs], gv);
    copy16_g2l(&sVt[buf][cr * LDK + cs + 8], gv + 8);
  };

  v16h qf0, qf1;
  {
    const _Float16* g = qbh + (size_t)(q0 + lrow) * D_HEAD;
    qf0 = frag_ld(g, 0 + kofs);
    qf1 = frag_ld(g, 32 + kofs);
  }

  float mrow[8], lsum[8];
#pragma unroll
  for (int r = 0; r < 8; ++r) { mrow[r] = -1e30f; lsum[r] = 0.f; }
  v8f o[4];
#pragma unroll
  for (int j = 0; j < 4; ++j) o[j] = (v8f){0,0,0,0,0,0,0,0};

  load_chunk(0, 0);
  async_wait_all();
  __syncthreads();

  for (int c0 = 0; c0 < SEQ; c0 += SK) {
    const int cur = (c0 / SK) & 1;
    if (c0 + SK < SEQ) load_chunk(cur ^ 1, c0 + SK);

    // S = (Q . K^T) * scale  -> 4 C-fragments (16x64)
    v8f s[4];
#pragma unroll
    for (int j = 0; j < 4; ++j) {
      s[j] = (v8f){0,0,0,0,0,0,0,0};
      const _Float16* kb = &sK[cur][(j * 16 + lrow) * LDK];
      v16h bf0 = frag_ld(kb, 0 + kofs);
      v16h bf1 = frag_ld(kb, 32 + kofs);
      s[j] = wmma_f16(qf0, bf0, s[j]);
      s[j] = wmma_f16(qf1, bf1, s[j]);
    }
    const float scale = 0.125f;   // 1/sqrt(64)
#pragma unroll
    for (int j = 0; j < 4; ++j)
#pragma unroll
      for (int r = 0; r < 8; ++r) s[j][r] *= scale;

    // Online softmax: row M = fixed VGPR r on one 16-lane half
    float corr[8];
#pragma unroll
    for (int r = 0; r < 8; ++r) {
      float mj = fmaxf(fmaxf(s[0][r], s[1][r]), fmaxf(s[2][r], s[3][r]));
#pragma unroll
      for (int m = 1; m < 16; m <<= 1) mj = fmaxf(mj, __shfl_xor(mj, m, 32));
      float mnew = fmaxf(mrow[r], mj);
      corr[r] = __expf(mrow[r] - mnew);
      mrow[r] = mnew;
    }

    _Float16* pw = &sP[wid * 16 * LDK];
#pragma unroll
    for (int r = 0; r < 8; ++r) {
      float rs = 0.f;
#pragma unroll
      for (int j = 0; j < 4; ++j) {
        float p = __expf(s[j][r] - mrow[r]);
        rs += p;
        pw[(r + half8) * LDK + j * 16 + lrow] = (_Float16)p;
      }
#pragma unroll
      for (int m = 1; m < 16; m <<= 1) rs += __shfl_xor(rs, m, 32);
      lsum[r] = lsum[r] * corr[r] + rs;
    }
#pragma unroll
    for (int j = 0; j < 4; ++j)
#pragma unroll
      for (int r = 0; r < 8; ++r) o[j][r] *= corr[r];

    // P (A-layout via per-wave LDS) x V fragments
    v16h pf0 = frag_ld(&pw[lrow * LDK], 0 + kofs);
    v16h pf1 = frag_ld(&pw[lrow * LDK], 32 + kofs);
#pragma unroll
    for (int j = 0; j < 4; ++j) {
      const _Float16* vb = &sVt[cur][(j * 16 + lrow) * LDK];
      v16h vf0 = frag_ld(vb, 0 + kofs);
      v16h vf1 = frag_ld(vb, 32 + kofs);
      o[j] = wmma_f16(pf0, vf0, o[j]);
      o[j] = wmma_f16(pf1, vf1, o[j]);
    }
    async_wait_all();
    __syncthreads();
  }

  const int b = bh >> 4, h = bh & 15;
#pragma unroll
  for (int j = 0; j < 4; ++j)
#pragma unroll
    for (int r = 0; r < 8; ++r) {
      int row = q0 + r + half8;
      size_t m = (size_t)b * SEQ + row;
      int col = h * D_HEAD + j * 16 + lrow;
      O[m * D_MODEL + col] = (_Float16)(o[j][r] / lsum[r]);
    }
}

// ---------------------------------------------------------------------------
// Output projection: [8192,1024] x Wt[1024,1024] + bias -> fp32 d_out
// ---------------------------------------------------------------------------
__launch_bounds__(256)
__global__ void out_gemm(const _Float16* __restrict__ A,    // [8192][1024]
                         const _Float16* __restrict__ Wt,   // [1024][1024] (N-major)
                         const float*    __restrict__ bias, // [1024]
                         float* __restrict__ out) {
  __shared__ _Float16 sA[2][BM * LDT];
  __shared__ _Float16 sB[2][BN * LDT];

  const int tid  = threadIdx.x;
  const int lane = tid & 31;
  const int wid  = tid >> 5;
  const int wm   = wid & 3;
  const int wn   = wid >> 2;
  const int m0   = blockIdx.y * BM;
  const int n0   = blockIdx.x * BN;
  const int lrow = lane & 15;
  const int kofs = (lane < 16) ? 0 : 8;
  const int half8 = (lane >> 4) << 3;

  const int cr = tid >> 1;
  const int cs = (tid & 1) * 32;
  auto load_tiles = [&](int buf, int k0) {
    const _Float16* ga = A + (size_t)(m0 + cr) * D_MODEL + k0 + cs;
    _Float16* la = &sA[buf][cr * LDT + cs];
#pragma unroll
    for (int p = 0; p < 4; ++p) copy16_g2l(la + p * 8, ga + p * 8);
    const _Float16* gb = Wt + (size_t)(n0 + cr) * D_MODEL + k0 + cs;
    _Float16* lb = &sB[buf][cr * LDT + cs];
#pragma unroll
    for (int p = 0; p < 4; ++p) copy16_g2l(lb + p * 8, gb + p * 8);
  };

  v8f acc[2][4];
#pragma unroll
  for (int i = 0; i < 2; ++i)
#pragma unroll
    for (int j = 0; j < 4; ++j) acc[i][j] = (v8f){0,0,0,0,0,0,0,0};

  load_tiles(0, 0);
  async_wait_all();
  __syncthreads();

  const int NK = D_MODEL / BK;   // 16
  for (int kk = 0; kk < NK; ++kk) {
    const int cur = kk & 1;
    if (kk + 1 < NK) load_tiles(cur ^ 1, (kk + 1) * BK);

#pragma unroll
    for (int kc = 0; kc < 2; ++kc) {
      const int ko = kc * 32 + kofs;
      v16h af0 = frag_ld(&sA[cur][(wm * 32 +      lrow) * LDT], ko);
      v16h af1 = frag_ld(&sA[cur][(wm * 32 + 16 + lrow) * LDT], ko);
#pragma unroll
      for (int j = 0; j < 4; ++j) {
        v16h bf = frag_ld(&sB[cur][(wn * 64 + j * 16 + lrow) * LDT], ko);
        acc[0][j] = wmma_f16(af0, bf, acc[0][j]);
        acc[1][j] = wmma_f16(af1, bf, acc[1][j]);
      }
    }
    async_wait_all();
    __syncthreads();
  }

#pragma unroll
  for (int i = 0; i < 2; ++i)
#pragma unroll
    for (int j = 0; j < 4; ++j)
#pragma unroll
      for (int r = 0; r < 8; ++r) {
        int m = m0 + wm * 32 + i * 16 + r + half8;
        int c = n0 + wn * 64 + j * 16 + lrow;
        out[(size_t)m * D_MODEL + c] = acc[i][j][r] + bias[c];
      }
}

// ---------------------------------------------------------------------------
extern "C" void kernel_launch(void* const* d_in, const int* in_sizes, int n_in,
                              void* d_out, int out_size, void* d_ws,
                              size_t ws_size, hipStream_t stream) {
  const float* x     = (const float*)d_in[0];
  const float* w_qkv = (const float*)d_in[1];
  const float* b_qkv = (const float*)d_in[2];
  const float* w_out = (const float*)d_in[3];
  const float* b_out = (const float*)d_in[4];

  char* ws = (char*)d_ws;
  size_t off = 0;
  auto wsalloc = [&](size_t bytes) -> void* {
    void* p = ws + off;
    off += (bytes + 255) & ~(size_t)255;
    return p;
  };
  const size_t nX  = (size_t)M_TOTAL * D_MODEL;
  const size_t nWq = (size_t)D_MODEL * N_QKV;
  const size_t nWo = (size_t)D_MODEL * D_MODEL;
  _Float16* xh  = (_Float16*)wsalloc(nX * 2);
  _Float16* wqt = (_Float16*)wsalloc(nWq * 2);   // [3072][1024]
  _Float16* wot = (_Float16*)wsalloc(nWo * 2);   // [1024][1024] N-major
  _Float16* qb  = (_Float16*)wsalloc(nX * 2);
  _Float16* kb  = (_Float16*)wsalloc(nX * 2);
  _Float16* vb  = (_Float16*)wsalloc(nX * 2);    // [B,H,Dh,T]
  _Float16* ao  = (_Float16*)wsalloc(nX * 2);

  const int TPB = 256;
  cast_f32_to_f16_v8<<<(unsigned)((nX / 8 + TPB - 1) / TPB), TPB, 0, stream>>>(
      x, xh, (int)(nX / 8));
  cast_transpose_f16<<<dim3(N_QKV / 32, D_MODEL / 32), dim3(32, 8), 0, stream>>>(
      w_qkv, wqt, D_MODEL, N_QKV);
  cast_transpose_f16<<<dim3(D_MODEL / 32, D_MODEL / 32), dim3(32, 8), 0, stream>>>(
      w_out, wot, D_MODEL, D_MODEL);

  qkv_gemm<<<dim3(N_QKV / BN, M_TOTAL / BM), 256, 0, stream>>>(
      xh, wqt, b_qkv, qb, kb, vb);

  flash_attn<<<dim3(SEQ / 128, BATCH * N_HEAD), 256, 0, stream>>>(
      qb, kb, vb, ao);

  out_gemm<<<dim3(D_MODEL / BN, M_TOTAL / BM), 256, 0, stream>>>(
      ao, wot, b_out, (float*)d_out);
}
